// LinearAttention_82875688943695
// MI455X (gfx1250) — matrix operand order repaired and checked by reading, not compiled, not run on hardware
//
#include <hip/hip_runtime.h>
#include <hip/hip_bf16.h>

// ---------------------------------------------------------------------------
// LinearAttention fused pipeline for MI455X (gfx1250), wave32.
//   x:(16,256,64,64) w_qkv:(384,256) w_out:(256,128) b_out:(256) g:(256)
// Big GEMMs use f16 WMMA 16x16x32 (8x K-depth of the f32 op -> compute stays
// under the 23.3 TB/s bandwidth floor); small reductions stay f32 WMMA.
// q-tile staging in kernel 4 uses gfx1250 async global->LDS copies.
// ---------------------------------------------------------------------------

typedef __attribute__((ext_vector_type(2)))  float    v2f;
typedef __attribute__((ext_vector_type(8)))  float    v8f;
typedef __attribute__((ext_vector_type(2)))  _Float16 v2h;
typedef __attribute__((ext_vector_type(8)))  _Float16 v8h;
typedef __attribute__((ext_vector_type(16))) _Float16 v16h;

#define BATCH   16
#define CIN     256
#define NPOS    4096
#define QKVC    384
#define HEADS   4
#define DH      32
#define HID     128
#define OC      256
#define SCALE   0.17677669529663689f   // 32^-0.5

__device__ __forceinline__ v8f wmma4(v2f a, v2f b, v8f c) {
  return __builtin_amdgcn_wmma_f32_16x16x4_f32(false, a, false, b, (short)0, c,
                                               false, false);
}
__device__ __forceinline__ v8f wmma16(v16h a, v16h b, v8f c) {
  return __builtin_amdgcn_wmma_f32_16x16x32_f16(false, a, false, b, (short)0, c,
                                                false, false);
}
__device__ __forceinline__ v16h combine8(v8h lo, v8h hi) {
  return __builtin_shufflevector(lo, hi, 0, 1, 2, 3, 4, 5, 6, 7, 8, 9, 10, 11,
                                 12, 13, 14, 15);
}
__device__ __forceinline__ v8h cvt8(float4 a, float4 b) {
  v8h h = {(_Float16)a.x, (_Float16)a.y, (_Float16)a.z, (_Float16)a.w,
           (_Float16)b.x, (_Float16)b.y, (_Float16)b.z, (_Float16)b.w};
  return h;
}
// gfx1250 async global->LDS copy (ASYNCcnt-tracked). Generic __shared__
// pointers carry the wave-relative LDS byte address in their low 32 bits.
__device__ __forceinline__ void async_copy_b128(const void* gsrc, void* lds) {
  unsigned lds_off = (unsigned)(size_t)lds;
  unsigned long long ga = (unsigned long long)(size_t)gsrc;
  asm volatile("global_load_async_to_lds_b128 %0, %1, off" ::"v"(lds_off),
               "v"(ga)
               : "memory");
}
__device__ __forceinline__ void wait_async_zero() {
  asm volatile("s_wait_asynccnt 0" ::: "memory");
}

// ===========================================================================
// Kernel 1: qkv[b, o, p] = sum_c w_qkv[o, c] * x[b, c, p]     (f16 WMMA)
// Block tile M=64 x N=128, K chunks of 64 staged into LDS as f16,
// K-contiguous per row: As[m][k], Bs[n][k] (rows padded to 72 halves=144B).
// 8 waves in 2(m) x 4(n); each wave: 2x2 tiles of 16x16, 32 WMMA total.
// ===========================================================================
__global__ __launch_bounds__(256) void qkv_gemm_kernel(
    const float* __restrict__ x, const float* __restrict__ w,
    float* __restrict__ qkv) {
  const int b    = blockIdx.z;
  const int oblk = blockIdx.y * 64;
  const int pblk = blockIdx.x * 128;
  const int tid  = threadIdx.x;
  const int lane = tid & 31;
  const int wid  = tid >> 5;
  const int wm   = wid >> 2;
  const int wn   = wid & 3;
  const int lr   = lane & 15;
  const int lh   = lane >> 4;

  __shared__ _Float16 As[64][72];    // [m][k], row = 144B (16B multiple)
  __shared__ _Float16 Bs[128][72];   // [n][k] (transposed from x)

  const float* xb = x + (size_t)b * CIN * NPOS;
  v8f acc[2][2] = {};

  for (int kb = 0; kb < CIN; kb += 64) {
    __syncthreads();
    {  // A: W[oblk+m][kb..kb+63] -> As[m][*] (cvt f32->f16)
      const int m  = tid >> 2;
      const int kq = (tid & 3) * 16;
      const float* src = w + (size_t)(oblk + m) * CIN + kb + kq;
      const float4 f0 = *(const float4*)(src + 0);
      const float4 f1 = *(const float4*)(src + 4);
      const float4 f2 = *(const float4*)(src + 8);
      const float4 f3 = *(const float4*)(src + 12);
      *(v8h*)&As[m][kq + 0] = cvt8(f0, f1);
      *(v8h*)&As[m][kq + 8] = cvt8(f2, f3);
    }
    {  // B: x[kb+k..k+1][pblk+n4..n4+3] -> Bs[n][k..k+1] (transpose + cvt)
#pragma unroll
      for (int it = 0; it < 4; ++it) {
        const int qd = tid + it * 256;        // 0..1023
        const int n4 = (qd & 31) * 4;         // 0..124
        const int k  = (qd >> 5) * 2;         // 0..62
        const float4 f = *(const float4*)(xb + (size_t)(kb + k) * NPOS + pblk + n4);
        const float4 g = *(const float4*)(xb + (size_t)(kb + k + 1) * NPOS + pblk + n4);
        *(v2h*)&Bs[n4 + 0][k] = (v2h){(_Float16)f.x, (_Float16)g.x};
        *(v2h*)&Bs[n4 + 1][k] = (v2h){(_Float16)f.y, (_Float16)g.y};
        *(v2h*)&Bs[n4 + 2][k] = (v2h){(_Float16)f.z, (_Float16)g.z};
        *(v2h*)&Bs[n4 + 3][k] = (v2h){(_Float16)f.w, (_Float16)g.w};
      }
    }
    __syncthreads();

#pragma unroll
    for (int ks = 0; ks < 64; ks += 32) {
      // A frag: per lane two contiguous 8-half runs at K = 8*lh and 16+8*lh.
      const _Float16* ap0 = &As[wm * 32 + lr][ks + 8 * lh];
      const _Float16* ap1 = &As[wm * 32 + 16 + lr][ks + 8 * lh];
      const v16h a0 = combine8(*(const v8h*)ap0, *(const v8h*)(ap0 + 16));
      const v16h a1 = combine8(*(const v8h*)ap1, *(const v8h*)(ap1 + 16));
      // B frag: per lane two contiguous 8-half runs at K = 8*lh and 16+8*lh.
      const _Float16* bp0 = &Bs[wn * 32 + lr][ks + 8 * lh];
      const _Float16* bp1 = &Bs[wn * 32 + 16 + lr][ks + 8 * lh];
      const v16h b0 = combine8(*(const v8h*)bp0, *(const v8h*)(bp0 + 16));
      const v16h b1 = combine8(*(const v8h*)bp1, *(const v8h*)(bp1 + 16));
      acc[0][0] = wmma16(a0, b0, acc[0][0]);
      acc[0][1] = wmma16(a0, b1, acc[0][1]);
      acc[1][0] = wmma16(a1, b0, acc[1][0]);
      acc[1][1] = wmma16(a1, b1, acc[1][1]);
    }
  }

  float* outb = qkv + (size_t)b * QKVC * NPOS;
#pragma unroll
  for (int mt = 0; mt < 2; ++mt)
#pragma unroll
    for (int nt = 0; nt < 2; ++nt)
#pragma unroll
      for (int r = 0; r < 8; ++r) {
        const int o = oblk + wm * 32 + mt * 16 + r + 8 * lh;
        const int p = pblk + wn * 32 + nt * 16 + lr;
        outb[(size_t)o * NPOS + p] = acc[mt][nt][r];
      }
}

// ===========================================================================
// Kernel 1b: one-time w_out f32 -> f16 conversion (256x128 elements).
// ===========================================================================
typedef __attribute__((ext_vector_type(4))) _Float16 v4h;
__global__ __launch_bounds__(256) void wcvt_kernel(const float* __restrict__ w,
                                                   _Float16* __restrict__ wh) {
  const int i = (blockIdx.x * 256 + threadIdx.x) * 4;
  const float4 v = *(const float4*)(w + i);
  v4h h = {(_Float16)v.x, (_Float16)v.y, (_Float16)v.z, (_Float16)v.w};
  *(v4h*)(wh + i) = h;
}

// ===========================================================================
// Kernel 2: per k-row (b, h*32+d): max over n, sumexp over n.
// ===========================================================================
__global__ __launch_bounds__(256) void kstats_kernel(
    const float* __restrict__ qkv, float* __restrict__ kmax,
    float* __restrict__ ksum) {
  const int row = blockIdx.x;
  const int b   = row >> 7;
  const int ch  = row & 127;
  const float* kr = qkv + (size_t)b * QKVC * NPOS + (size_t)(HID + ch) * NPOS;
  const int tid = threadIdx.x;
  __shared__ float red[256];

  float m = -3.0e38f;
  for (int i = tid; i < NPOS; i += 256) m = fmaxf(m, kr[i]);
  red[tid] = m;
  __syncthreads();
  for (int s = 128; s > 0; s >>= 1) {
    if (tid < s) red[tid] = fmaxf(red[tid], red[tid + s]);
    __syncthreads();
  }
  m = red[0];
  __syncthreads();

  float ssum = 0.f;
  for (int i = tid; i < NPOS; i += 256) ssum += __expf(kr[i] - m);
  red[tid] = ssum;
  __syncthreads();
  for (int s = 128; s > 0; s >>= 1) {
    if (tid < s) red[tid] += red[tid + s];
    __syncthreads();
  }
  if (tid == 0) {
    kmax[row] = m;
    ksum[row] = red[0];
  }
}

// ===========================================================================
// Kernel 3: context[b,h,d,e] = sum_n softmax_n(k)[d,n] * v[e,n] / NPOS.
// f32 WMMA with exp fused into the A fragment; cross-wave LDS reduction.
// ===========================================================================
__global__ __launch_bounds__(256) void context_kernel(
    const float* __restrict__ qkv, const float* __restrict__ kmax,
    const float* __restrict__ ksum, float* __restrict__ ctx) {
  const int bh = blockIdx.x;
  const int b  = bh >> 2;
  const int h  = bh & 3;
  const float* kbase = qkv + (size_t)b * QKVC * NPOS + (size_t)(HID + h * DH) * NPOS;
  const float* vbase = qkv + (size_t)b * QKVC * NPOS + (size_t)(2 * HID + h * DH) * NPOS;
  const int tid  = threadIdx.x;
  const int lane = tid & 31;
  const int wid  = tid >> 5;
  const int lr   = lane & 15;
  const int lh   = lane >> 4;

  const float km0 = kmax[bh * DH + lr];
  const float km1 = kmax[bh * DH + 16 + lr];

  v8f acc[2][2] = {};
  const int nbase = wid * 512;
  for (int nn = 0; nn < 512; nn += 4) {
    const int na = nbase + nn + 2 * lh;
    v2f a0 = {__expf(kbase[(size_t)lr * NPOS + na] - km0),
              __expf(kbase[(size_t)lr * NPOS + na + 1] - km0)};
    v2f a1 = {__expf(kbase[(size_t)(16 + lr) * NPOS + na] - km1),
              __expf(kbase[(size_t)(16 + lr) * NPOS + na + 1] - km1)};
    v2f b0 = {vbase[(size_t)lr * NPOS + na], vbase[(size_t)lr * NPOS + na + 1]};
    v2f b1 = {vbase[(size_t)(16 + lr) * NPOS + na],
              vbase[(size_t)(16 + lr) * NPOS + na + 1]};
    acc[0][0] = wmma4(a0, b0, acc[0][0]);
    acc[0][1] = wmma4(a0, b1, acc[0][1]);
    acc[1][0] = wmma4(a1, b0, acc[1][0]);
    acc[1][1] = wmma4(a1, b1, acc[1][1]);
  }

  __shared__ float part[8][32][32];
#pragma unroll
  for (int mt = 0; mt < 2; ++mt)
#pragma unroll
    for (int nt = 0; nt < 2; ++nt)
#pragma unroll
      for (int r = 0; r < 8; ++r)
        part[wid][mt * 16 + r + 8 * lh][nt * 16 + lr] = acc[mt][nt][r];
  __syncthreads();

  for (int i = tid; i < DH * DH; i += 256) {
    const int d = i >> 5;
    const int e = i & 31;
    float s = 0.f;
#pragma unroll
    for (int w2 = 0; w2 < 8; ++w2) s += part[w2][d][e];
    const float scale = 1.0f / (ksum[bh * DH + d] * (float)NPOS);
    ctx[(size_t)bh * DH * DH + d * DH + e] = s * scale;
  }
}

// ===========================================================================
// Kernel 4: per 32-position tile:
//   q_sm = softmax_d(q)*SCALE          (q staged via async global->LDS)
//   mid  = ctx^T @ q_sm                (f32 WMMA, K=32; stored f16 [p][e])
//   y    = W_out(f16) @ mid + b_out    (f16 WMMA, K=128)
//   LayerNorm over 256 channels.
// ===========================================================================
__global__ __launch_bounds__(256) void fused_out_kernel(
    const float* __restrict__ qkv, const float* __restrict__ ctx,
    const _Float16* __restrict__ wouth, const float* __restrict__ bout,
    const float* __restrict__ g, float* __restrict__ out) {
  const int b  = blockIdx.y;
  const int p0 = blockIdx.x * 32;
  const int tid  = threadIdx.x;
  const int lane = tid & 31;
  const int wid  = tid >> 5;
  const int lr   = lane & 15;
  const int lh   = lane >> 4;

  __shared__ float smem[4096 + 4096 + 512];
  __shared__ _Float16 midh[32][136];   // [p][e], row = 272B (16B multiple)
  float* ctxs = smem;                  // [h][d][e] 4096
  float* qs   = smem + 4096;           // [ch][p]   4096
  float* red  = smem + 8192;           // LN scratch 512
  float* ys   = smem;                  // [o][p] 8192 floats, aliases ctxs+qs

  const float* qbase = qkv + (size_t)b * QKVC * NPOS;

  // ---- Phase A: stage ctx (sync) + q tile (async b128), softmax over d ---
  for (int i = tid; i < HEADS * DH * DH; i += 256)
    ctxs[i] = ctx[(size_t)b * HEADS * DH * DH + i];
#pragma unroll
  for (int it = 0; it < 4; ++it) {
    const int qd = tid + it * 256;     // 0..1023 quads of 4 floats
    const int ch = qd >> 3;
    const int pq = (qd & 7) * 4;
    async_copy_b128(qbase + (size_t)ch * NPOS + p0 + pq, &qs[ch * 32 + pq]);
  }
  wait_async_zero();
  __syncthreads();

  if (tid < HEADS * 32) {
    const int h = tid >> 5;
    const int p = tid & 31;
    float mx = -3.0e38f;
#pragma unroll
    for (int d = 0; d < DH; ++d) mx = fmaxf(mx, qs[(h * DH + d) * 32 + p]);
    float s = 0.f;
#pragma unroll
    for (int d = 0; d < DH; ++d) {
      const float ev = __expf(qs[(h * DH + d) * 32 + p] - mx);
      qs[(h * DH + d) * 32 + p] = ev;
      s += ev;
    }
    const float inv = SCALE / s;
#pragma unroll
    for (int d = 0; d < DH; ++d) qs[(h * DH + d) * 32 + p] *= inv;
  }
  __syncthreads();

  // ---- mid[e, p] = sum_d ctx[h][d][e%32] * q_sm[h*32+d][p]  (f32 WMMA) ---
  {
    const int mt = wid;
    const int h  = mt >> 1;
    const int el = (mt & 1) * 16 + lr;
    v8f m0 = {}, m1 = {};
#pragma unroll
    for (int kc = 0; kc < DH; kc += 4) {
      const int d = kc + 2 * lh;
      v2f a  = {ctxs[h * 1024 + d * DH + el], ctxs[h * 1024 + (d + 1) * DH + el]};
      v2f b0 = {qs[(h * DH + d) * 32 + lr], qs[(h * DH + d + 1) * 32 + lr]};
      v2f b1 = {qs[(h * DH + d) * 32 + 16 + lr],
                qs[(h * DH + d + 1) * 32 + 16 + lr]};
      m0 = wmma4(a, b0, m0);
      m1 = wmma4(a, b1, m1);
    }
#pragma unroll
    for (int r = 0; r < 8; ++r) {
      const int e = mt * 16 + r + 8 * lh;
      midh[lr][e]      = (_Float16)m0[r];
      midh[16 + lr][e] = (_Float16)m1[r];
    }
  }
  __syncthreads();

  // ---- Phase B: y[o, p] = sum_c wout[o, c] * mid[c, p] + bout[o] (f16) ---
  {
    v8f acc[2][2] = {};
#pragma unroll
    for (int kb = 0; kb < HID; kb += 32) {
      const int o0 = wid * 32 + lr;
      const _Float16* ap0 = wouth + (size_t)o0 * HID + kb + 8 * lh;
      const _Float16* ap1 = wouth + (size_t)(o0 + 16) * HID + kb + 8 * lh;
      const v16h a0 = combine8(*(const v8h*)ap0, *(const v8h*)(ap0 + 16));
      const v16h a1 = combine8(*(const v8h*)ap1, *(const v8h*)(ap1 + 16));
      const _Float16* bp0 = &midh[lr][kb + 8 * lh];
      const _Float16* bp1 = &midh[16 + lr][kb + 8 * lh];
      const v16h b0 = combine8(*(const v8h*)bp0, *(const v8h*)(bp0 + 16));
      const v16h b1 = combine8(*(const v8h*)bp1, *(const v8h*)(bp1 + 16));
      acc[0][0] = wmma16(a0, b0, acc[0][0]);
      acc[0][1] = wmma16(a0, b1, acc[0][1]);
      acc[1][0] = wmma16(a1, b0, acc[1][0]);
      acc[1][1] = wmma16(a1, b1, acc[1][1]);
    }
#pragma unroll
    for (int mt = 0; mt < 2; ++mt)
#pragma unroll
      for (int nt = 0; nt < 2; ++nt)
#pragma unroll
        for (int r = 0; r < 8; ++r) {
          const int o = wid * 32 + mt * 16 + r + 8 * lh;
          const int p = nt * 16 + lr;
          ys[o * 32 + p] = acc[mt][nt][r] + bout[o];
        }
  }
  __syncthreads();

  // ---- Phase C: LayerNorm over the 256 channels of each column ----------
  {
    const int col = tid & 31;
    const int seg = tid >> 5;
    float s = 0.f, s2 = 0.f;
#pragma unroll
    for (int r = 0; r < 32; ++r) {
      const float v = ys[(seg * 32 + r) * 32 + col];
      s += v;
      s2 += v * v;
    }
    red[seg * 32 + col]       = s;
    red[256 + seg * 32 + col] = s2;
    __syncthreads();
    if (seg == 0) {
      float ts = 0.f, ts2 = 0.f;
#pragma unroll
      for (int q2 = 0; q2 < 8; ++q2) {
        ts  += red[q2 * 32 + col];
        ts2 += red[256 + q2 * 32 + col];
      }
      const float mn  = ts * (1.0f / 256.0f);
      const float var = ts2 * (1.0f / 256.0f) - mn * mn;
      red[col]      = mn;
      red[32 + col] = rsqrtf(var + 1e-5f);
    }
    __syncthreads();
    const float mn = red[col];
    const float rs = red[32 + col];
    float* outp = out + (size_t)b * OC * NPOS + p0 + col;
#pragma unroll
    for (int r = 0; r < 32; ++r) {
      const int o = seg * 32 + r;
      outp[(size_t)o * NPOS] = (ys[o * 32 + col] - mn) * rs * g[o];
    }
  }
}

// ===========================================================================
extern "C" void kernel_launch(void* const* d_in, const int* in_sizes, int n_in,
                              void* d_out, int out_size, void* d_ws,
                              size_t ws_size, hipStream_t stream) {
  (void)in_sizes; (void)n_in; (void)out_size; (void)ws_size;
  const float* x     = (const float*)d_in[0];
  const float* w_qkv = (const float*)d_in[1];
  const float* w_out = (const float*)d_in[2];
  const float* b_out = (const float*)d_in[3];
  const float* g     = (const float*)d_in[4];
  float* out = (float*)d_out;

  float* ws        = (float*)d_ws;
  float* qkv       = ws;                               // 16*384*4096 f32
  float* kmax      = ws + (size_t)BATCH * QKVC * NPOS; // 2048 f32
  float* ksum      = kmax + BATCH * HID;               // 2048 f32
  float* ctx       = ksum + BATCH * HID;               // 16*4*32*32 f32
  _Float16* wouth  = (_Float16*)(ctx + BATCH * HEADS * DH * DH); // 32768 f16

  wcvt_kernel<<<OC * HID / 1024, 256, 0, stream>>>(w_out, wouth);
  qkv_gemm_kernel<<<dim3(NPOS / 128, QKVC / 64, BATCH), 256, 0, stream>>>(
      x, w_qkv, qkv);
  kstats_kernel<<<BATCH * HID, 256, 0, stream>>>(qkv, kmax, ksum);
  context_kernel<<<BATCH * HEADS, 256, 0, stream>>>(qkv, kmax, ksum, ctx);
  fused_out_kernel<<<dim3(NPOS / 32, BATCH), 256, 0, stream>>>(
      qkv, ctx, wouth, b_out, g, out);
}
